// VOnlyPruner_3659312136469
// MI455X (gfx1250) — matrix operand order repaired and checked by reading, not compiled, not run on hardware
//
#include <hip/hip_runtime.h>

// Problem constants (from reference): B=2,H=16,T=2048,D=64
#define BH      32
#define T       2048
#define D       64
#define KKEEP   782          // max(64, int(2048*0.382))
#define KPAD    800          // KKEEP padded to multiple of 32 (zero-filled)
#define RECENCY 64
#define QTILES  (T / 16)

typedef __attribute__((ext_vector_type(16))) __bf16 v16bf;
typedef __attribute__((ext_vector_type(8)))  float  v8f;

static __device__ __forceinline__ v8f wmma_bf16(v16bf a, v16bf b, v8f c) {
  // D(f32 16x16) = A(bf16 16x32) * B(bf16 32x16) + C
  return __builtin_amdgcn_wmma_f32_16x16x32_bf16(false, a, false, b, (short)0, c, false, false);
}

// ---- WMMA fragment loaders (this file's contiguous k order, wave32) -----------------
// A (16x32, 16-bit): lane L holds row m=L&15; lanes 0-15 cover K=0..15,
// lanes 16-31 cover K=16..31.  Element i -> K = kbase + 16*(L>=16) + i.
static __device__ __forceinline__ v16bf load_a_f32(const float* base, int ld,
                                                   int lane, int kchunk) {
  int m = lane & 15, half = lane >> 4;
  const float* row = base + (size_t)m * ld + kchunk + (half ? 16 : 0);
  v16bf a;
#pragma unroll
  for (int i = 0; i < 16; ++i) {
    int off = i;
    a[i] = (__bf16)row[off];
  }
  return a;
}

// B (32x16, 16-bit), built as the transpose of a row-major [key][d] tensor:
// lane L holds column n=L&15; element i -> Kdim = 16*(L>=16) + i.
static __device__ __forceinline__ v16bf load_bt_bf16(const __bf16* base, int ld,
                                                     int lane, int nbase, int kchunk) {
  int n = lane & 15, half = lane >> 4;
  const __bf16* row = base + (size_t)(nbase + n) * ld + kchunk + (half ? 16 : 0);
  v16bf b;
#pragma unroll
  for (int i = 0; i < 16; ++i) b[i] = row[i];
  return b;
}

// 16-lane reductions (xor masks < 16 stay within each half of the wave)
static __device__ __forceinline__ float redmax16(float x) {
#pragma unroll
  for (int off = 1; off < 16; off <<= 1) x = fmaxf(x, __shfl_xor(x, off, 32));
  return x;
}
static __device__ __forceinline__ float redsum16(float x) {
#pragma unroll
  for (int off = 1; off < 16; off <<= 1) x += __shfl_xor(x, off, 32);
  return x;
}

// ---- K0a: zero importance accumulator --------------------------------------
__global__ void k_zero(float* __restrict__ p, int n) {
  int i = blockIdx.x * blockDim.x + threadIdx.x;
  if (i < n) p[i] = 0.f;
}

// ---- K0b: one-time fp32 -> bf16 conversion of K (halves streamed bytes) ----
__global__ __launch_bounds__(256)
void k_prep(const float* __restrict__ k, __bf16* __restrict__ kb16) {
  size_t i = ((size_t)blockIdx.x * 256 + threadIdx.x) * 4;
  if (i + 3 < (size_t)BH * T * D) {
#pragma unroll
    for (int j = 0; j < 4; ++j) kb16[i + j] = (__bf16)k[i + j];
  }
}

// ---- K1: fused softmax stats + column-mass importance -----------------------
// Three passes over the 128 key tiles; NO cross-lane shuffles in the hot
// loops (per-lane partials only): loop body = 2 bf16 b128 loads + 2 WMMA + exp.
__global__ __launch_bounds__(32)
void k_importance(const float* __restrict__ q, const __bf16* __restrict__ kb16,
                  const float* __restrict__ scale_p,
                  float* __restrict__ m_out, float* __restrict__ l_out,
                  float* __restrict__ imp) {
  const float scale = scale_p[0];
  int blk = blockIdx.x;
  int qt = blk % QTILES, bh = blk / QTILES;
  const float*  qb = q    + ((size_t)bh * T + (size_t)qt * 16) * D;
  const __bf16* kb = kb16 + (size_t)bh * T * D;
  int lane = threadIdx.x, half = lane >> 4, n = lane & 15;

  v16bf a0 = load_a_f32(qb, D, lane, 0);
  v16bf a1 = load_a_f32(qb, D, lane, 32);

  // ---- Pass A: per-lane running max; single reduction at the end ----
  float pmax[8];
#pragma unroll
  for (int r = 0; r < 8; ++r) pmax[r] = -3.0e38f;
  for (int kt = 0; kt < T / 16; ++kt) {
    __builtin_prefetch(kb + (size_t)((kt + 1) * 16 + n) * D, 0, 3); // next tile
    v16bf b0 = load_bt_bf16(kb, D, lane, kt * 16, 0);
    v16bf b1 = load_bt_bf16(kb, D, lane, kt * 16, 32);
    v8f c = {};
    c = wmma_bf16(a0, b0, c);
    c = wmma_bf16(a1, b1, c);
#pragma unroll
    for (int r = 0; r < 8; ++r) pmax[r] = fmaxf(pmax[r], c[r] * scale);
  }
  float mrow[8];
#pragma unroll
  for (int r = 0; r < 8; ++r) mrow[r] = redmax16(pmax[r]);

  // ---- Pass B: per-lane partial sum-exp with final m; reduce once ----
  float psum[8];
#pragma unroll
  for (int r = 0; r < 8; ++r) psum[r] = 0.f;
  for (int kt = 0; kt < T / 16; ++kt) {
    __builtin_prefetch(kb + (size_t)((kt + 1) * 16 + n) * D, 0, 3);
    v16bf b0 = load_bt_bf16(kb, D, lane, kt * 16, 0);
    v16bf b1 = load_bt_bf16(kb, D, lane, kt * 16, 32);
    v8f c = {};
    c = wmma_bf16(a0, b0, c);
    c = wmma_bf16(a1, b1, c);
#pragma unroll
    for (int r = 0; r < 8; ++r) psum[r] += __expf(c[r] * scale - mrow[r]);
  }
  float linv[8];
#pragma unroll
  for (int r = 0; r < 8; ++r) {
    float l = redsum16(psum[r]);
    psum[r] = l;
    linv[r] = 1.0f / l;
  }
  if (n == 0) {
#pragma unroll
    for (int r = 0; r < 8; ++r) {
      size_t row = (size_t)bh * T + (size_t)qt * 16 + r + 8 * half;
      m_out[row] = mrow[r];
      l_out[row] = psum[r];
    }
  }

  // ---- Pass C: column masses; 1 shuffle + 1 atomic per tile ----
  for (int kt = 0; kt < T / 16; ++kt) {
    __builtin_prefetch(kb + (size_t)((kt + 1) * 16 + n) * D, 0, 3);
    v16bf b0 = load_bt_bf16(kb, D, lane, kt * 16, 0);
    v16bf b1 = load_bt_bf16(kb, D, lane, kt * 16, 32);
    v8f c = {};
    c = wmma_bf16(a0, b0, c);
    c = wmma_bf16(a1, b1, c);
    float cs = 0.f;
#pragma unroll
    for (int r = 0; r < 8; ++r)
      cs += __expf(c[r] * scale - mrow[r]) * linv[r];
    cs += __shfl_xor(cs, 16, 32);   // combine rows 0-7 and 8-15 for same column
    if (half == 0)
      atomicAdd(&imp[(size_t)bh * T + kt * 16 + n], cs);
  }
}

// ---- K2: exposure correction + recency forcing -----------------------------
__global__ void k_final_imp(float* __restrict__ imp) {
  int i = blockIdx.x * blockDim.x + threadIdx.x;
  if (i >= BH * T) return;
  int t = i & (T - 1);
  float vv = imp[i] / (float)(T - t);
  if (t >= T - RECENCY) vv = 1.0f;
  imp[i] = vv;
}

// ---- K3: top-k via bitonic sort (desc by value, asc index tie-break) -------
__global__ __launch_bounds__(256)
void k_topk(const float* __restrict__ imp, int* __restrict__ idx_out) {
  __shared__ float sv[T];
  __shared__ int   si[T];
  int bh = blockIdx.x, tid = threadIdx.x;
  for (int i = tid; i < T; i += 256) { sv[i] = imp[(size_t)bh * T + i]; si[i] = i; }
  __syncthreads();
  for (int ksz = 2; ksz <= T; ksz <<= 1) {
    for (int j = ksz >> 1; j > 0; j >>= 1) {
      for (int i = tid; i < T; i += 256) {
        int ixj = i ^ j;
        if (ixj > i) {
          bool desc = ((i & ksz) == 0);
          float va = sv[i], vb = sv[ixj];
          int ia = si[i], ib = si[ixj];
          bool aFirst = (va > vb) || (va == vb && ia < ib);
          if (desc ? !aFirst : aFirst) {
            sv[i] = vb; sv[ixj] = va; si[i] = ib; si[ixj] = ia;
          }
        }
      }
      __syncthreads();
    }
  }
  for (int p = tid; p < KKEEP; p += 256)
    idx_out[(size_t)bh * KKEEP + p] = si[p];
}

// ---- K4: gather K_keep/V_keep (bf16 ws copies, zero-padded to KPAD) --------
__global__ __launch_bounds__(256)
void k_gather(const __bf16* __restrict__ kb16, const float* __restrict__ v,
              const int* __restrict__ idx,
              __bf16* __restrict__ kkeep, __bf16* __restrict__ vkeep,
              float* __restrict__ vkeep_out) {
  int bh = blockIdx.x;
  for (int p = threadIdx.x; p < KPAD; p += 256) {
    __bf16* kr = kkeep + ((size_t)bh * KPAD + p) * D;
    __bf16* vr = vkeep + ((size_t)bh * KPAD + p) * D;
    if (p < KKEEP) {
      int key = idx[(size_t)bh * KKEEP + p];
      const __bf16* ks = kb16 + ((size_t)bh * T + key) * D;
      const float*  vs = v    + ((size_t)bh * T + key) * D;
      float* vo = vkeep_out + ((size_t)bh * KKEEP + p) * D;
#pragma unroll 4
      for (int d = 0; d < D; ++d) {
        kr[d] = ks[d];
        float vv = vs[d];
        vr[d] = (__bf16)vv;
        vo[d] = vv;
      }
    } else {
#pragma unroll 4
      for (int d = 0; d < D; ++d) { kr[d] = (__bf16)0.f; vr[d] = (__bf16)0.f; }
    }
  }
}

// ---- K5: gathered scores -> P (LDS) -> renorm -> attn_pruned + P@V_keep ----
__global__ __launch_bounds__(32)
void k_attend(const float* __restrict__ q, const float* __restrict__ scale_p,
              const float* __restrict__ m_in, const float* __restrict__ l_in,
              const __bf16* __restrict__ kkeep, const __bf16* __restrict__ vkeep,
              float* __restrict__ attn_out, float* __restrict__ out) {
  __shared__ float Pt[16 * KPAD];   // 51200 B
  __shared__ float rsum[16];
  const float scale = scale_p[0];
  int blk = blockIdx.x;
  int qt = blk % QTILES, bh = blk / QTILES;
  int lane = threadIdx.x, half = lane >> 4, n = lane & 15;
  const float*  qb  = q     + ((size_t)bh * T + (size_t)qt * 16) * D;
  const __bf16* kkb = kkeep + (size_t)bh * KPAD * D;
  const __bf16* vkb = vkeep + (size_t)bh * KPAD * D;

  v16bf a0 = load_a_f32(qb, D, lane, 0);
  v16bf a1 = load_a_f32(qb, D, lane, 32);
  float mrow[8], linv[8];
#pragma unroll
  for (int r = 0; r < 8; ++r) {
    size_t row = (size_t)bh * T + (size_t)qt * 16 + r + 8 * half;
    mrow[r] = m_in[row];
    linv[r] = 1.0f / l_in[row];
  }

  // Scores against gathered keys; stage un-renormalized P in LDS while
  // accumulating per-lane row-sum partials (reduced once at the end).
  float rpart[8];
#pragma unroll
  for (int r = 0; r < 8; ++r) rpart[r] = 0.f;
  for (int kt = 0; kt < KPAD / 16; ++kt) {
    __builtin_prefetch(kkb + (size_t)((kt + 1) * 16 + n) * D, 0, 3);
    v16bf b0 = load_bt_bf16(kkb, D, lane, kt * 16, 0);
    v16bf b1 = load_bt_bf16(kkb, D, lane, kt * 16, 32);
    v8f c = {};
    c = wmma_bf16(a0, b0, c);
    c = wmma_bf16(a1, b1, c);
    int col = kt * 16 + n;
#pragma unroll
    for (int r = 0; r < 8; ++r) {
      float p = __expf(c[r] * scale - mrow[r]) * linv[r];
      if (col >= KKEEP) p = 0.f;   // mask zero-padded keys
      rpart[r] += p;
      Pt[(r + 8 * half) * KPAD + col] = p;
    }
  }
  // Row renormalization factors (sum over kept columns + 1e-10)
#pragma unroll
  for (int r = 0; r < 8; ++r) rpart[r] = redsum16(rpart[r]);
  if (n == 0) {
#pragma unroll
    for (int r = 0; r < 8; ++r)
      rsum[r + 8 * half] = 1.0f / (rpart[r] + 1e-10f);
  }
  __syncthreads();

  // Write attn_pruned (renormalized); pure strided store stream, no divisions
  {
    float* ao = attn_out + ((size_t)bh * T + (size_t)qt * 16) * KKEEP;
    for (int r = 0; r < 16; ++r) {
      float rs = rsum[r];
      const float* ps = &Pt[r * KPAD];
      float* aor = ao + (size_t)r * KKEEP;
      for (int cidx = lane; cidx < KKEEP; cidx += 32)
        aor[cidx] = ps[cidx] * rs;
    }
  }
  __syncthreads();

  // out(16x64) = P_norm(16xKPAD) @ V_keep(KPADx64); renorm folded into A frag
  v8f acc0 = 0, acc1 = 0, acc2 = 0, acc3 = 0;
  int m = lane & 15;
  float invm = rsum[m];
  for (int kc = 0; kc < KPAD / 32; ++kc) {
    __builtin_prefetch(vkb + (size_t)((kc + 1) * 32 + half * 16) * D + n, 0, 3);
    const float* prow = &Pt[m * KPAD + kc * 32 + (half ? 16 : 0)];
    v16bf ap;
#pragma unroll
    for (int i = 0; i < 16; ++i) {
      int off = i;
      ap[i] = (__bf16)(prow[off] * invm);
    }
#pragma unroll
    for (int nt = 0; nt < 4; ++nt) {
      const __bf16* vb = vkb + (size_t)(kc * 32 + half * 16) * D + nt * 16 + n;
      v16bf bp;
#pragma unroll
      for (int i = 0; i < 16; ++i) bp[i] = vb[(size_t)i * D];
      v8f cc = (nt == 0) ? acc0 : (nt == 1) ? acc1 : (nt == 2) ? acc2 : acc3;
      cc = wmma_bf16(ap, bp, cc);
      if (nt == 0) acc0 = cc; else if (nt == 1) acc1 = cc;
      else if (nt == 2) acc2 = cc; else acc3 = cc;
    }
  }
#pragma unroll
  for (int nt = 0; nt < 4; ++nt) {
    v8f cc = (nt == 0) ? acc0 : (nt == 1) ? acc1 : (nt == 2) ? acc2 : acc3;
#pragma unroll
    for (int r = 0; r < 8; ++r)
      out[((size_t)bh * T + (size_t)qt * 16 + r + 8 * half) * D + nt * 16 + n] = cc[r];
  }
}

// ---- host-side launcher ----------------------------------------------------
extern "C" void kernel_launch(void* const* d_in, const int* in_sizes, int n_in,
                              void* d_out, int out_size, void* d_ws, size_t ws_size,
                              hipStream_t stream) {
  const float* q     = (const float*)d_in[0];
  const float* k     = (const float*)d_in[1];
  const float* v     = (const float*)d_in[2];
  const float* scale = (const float*)d_in[3];

  // Workspace (bytes): m | l | importance | K_keep(bf16) | V_keep(bf16) | K(bf16)
  char* ws = (char*)d_ws;
  float*  m_buf = (float*)(ws);
  float*  l_buf = (float*)(ws + (size_t)BH * T * 4);
  float*  imp   = (float*)(ws + (size_t)BH * T * 8);
  __bf16* kkeep = (__bf16*)(ws + (size_t)BH * T * 12);
  __bf16* vkeep = (__bf16*)(ws + (size_t)BH * T * 12 + (size_t)BH * KPAD * D * 2);
  __bf16* kb16  = (__bf16*)(ws + (size_t)BH * T * 12 + (size_t)BH * KPAD * D * 4);

  // Output layout (elements, return order): out | idx | attn_pruned | V_keep
  float* outp = (float*)d_out;
  int*   idxp = (int*)d_out + (size_t)BH * T * D;
  float* attp = (float*)d_out + (size_t)BH * T * D + (size_t)BH * KKEEP;
  float* vkp  = attp + (size_t)BH * T * KKEEP;

  k_zero      <<<(BH * T + 255) / 256, 256, 0, stream>>>(imp, BH * T);
  k_prep      <<<(BH * T * D / 4 + 255) / 256, 256, 0, stream>>>(k, kb16);
  k_importance<<<BH * QTILES, 32, 0, stream>>>(q, kb16, scale, m_buf, l_buf, imp);
  k_final_imp <<<(BH * T + 255) / 256, 256, 0, stream>>>(imp);
  k_topk      <<<BH, 256, 0, stream>>>(imp, idxp);
  k_gather    <<<BH, 256, 0, stream>>>(kb16, v, idxp, kkeep, vkeep, vkp);
  k_attend    <<<BH * QTILES, 32, 0, stream>>>(q, scale, m_buf, l_buf, kkeep, vkeep, attp, outp);
}
